// GraphSAGE_89309549953639
// MI455X (gfx1250) — matrix operand order repaired and checked by reading, not compiled, not run on hardware
//
#include <hip/hip_runtime.h>
#include <hip/hip_fp16.h>
#include <math.h>

typedef __attribute__((ext_vector_type(16))) _Float16 v16h;
typedef __attribute__((ext_vector_type(8)))  float    v8f;

#define N_NODES 50000
#define DIM     128
#define N_EDGES 800000

// workspace layout (bytes), all offsets 256-aligned
#define OFF_DEG 0
#define OFF_AGG 204800                       // after deg (50000*4 = 200000)
#define OFF_H1  (OFF_AGG + N_NODES*DIM*4)    // 25 804 800
#define OFF_W   (OFF_H1  + N_NODES*DIM*4)    // 51 404 800  (4 x 16384 halves follow)

// ---------------------------------------------------------------- weights->f16
__global__ void cvt_weights(const float* __restrict__ w1l, const float* __restrict__ w1r,
                            const float* __restrict__ w2l, const float* __restrict__ w2r,
                            _Float16* __restrict__ out) {
    int i = blockIdx.x * blockDim.x + threadIdx.x;   // 0..16383, launched exactly
    out[i]             = (_Float16)w1l[i];
    out[16384 + i]     = (_Float16)w1r[i];
    out[32768 + i]     = (_Float16)w2l[i];
    out[49152 + i]     = (_Float16)w2r[i];
}

// ---------------------------------------------------------------- degree count
__global__ void deg_count(const int* __restrict__ dst, float* __restrict__ deg) {
    int e = blockIdx.x * blockDim.x + threadIdx.x;
    if (e < N_EDGES) atomicAdd(&deg[dst[e]], 1.0f);
}

// ------------------------------------------------- edge scatter: wave per edge
__global__ void scatter_add(const float* __restrict__ feat,
                            const int* __restrict__ src, const int* __restrict__ dst,
                            float* __restrict__ agg) {
    int tid = blockIdx.x * blockDim.x + threadIdx.x;
    int e = tid >> 5;            // one wave32 per edge: 32 lanes * 4 floats = 128
    if (e >= N_EDGES) return;
    int q = (tid & 31) * 4;
    int s = src[e], d = dst[e];
    const float4 v = *(const float4*)(feat + (long)s * DIM + q);
    float* a = agg + (long)d * DIM + q;
    atomicAdd(a + 0, v.x);
    atomicAdd(a + 1, v.y);
    atomicAdd(a + 2, v.z);
    atomicAdd(a + 3, v.w);
}

// ------------------------------------------------- A-fragment, contiguous k order (k = 16*hi + s)
// lane L (hl=L&15, hi=L>>4): row = hl, halves 0..15 = K(ka..ka+15)
__device__ __forceinline__ v16h load_frag_a(const float* __restrict__ row, int ka, float s) {
    const float4 q0 = *(const float4*)(row + ka);
    const float4 q1 = *(const float4*)(row + ka + 4);
    const float4 q2 = *(const float4*)(row + ka + 8);
    const float4 q3 = *(const float4*)(row + ka + 12);
    v16h a;
    a[0]  = (_Float16)(q0.x * s); a[1]  = (_Float16)(q0.y * s);
    a[2]  = (_Float16)(q0.z * s); a[3]  = (_Float16)(q0.w * s);
    a[4]  = (_Float16)(q1.x * s); a[5]  = (_Float16)(q1.y * s);
    a[6]  = (_Float16)(q1.z * s); a[7]  = (_Float16)(q1.w * s);
    a[8]  = (_Float16)(q2.x * s); a[9]  = (_Float16)(q2.y * s);
    a[10] = (_Float16)(q2.z * s); a[11] = (_Float16)(q2.w * s);
    a[12] = (_Float16)(q3.x * s); a[13] = (_Float16)(q3.y * s);
    a[14] = (_Float16)(q3.z * s); a[15] = (_Float16)(q3.w * s);
    return a;
}

// ---------------- fused SAGE layer GEMM: out = relu?( agg/deg @ Wl^T + X @ Wr^T + b )
// one wave per 16x16 output tile; N_NODES = 3125*16 exactly -> EXEC all-ones
__global__ void sage_gemm(const float* __restrict__ agg, const float* __restrict__ deg,
                          const float* __restrict__ xin,
                          const _Float16* __restrict__ wl, const _Float16* __restrict__ wr,
                          const float* __restrict__ bias, float* __restrict__ out,
                          int do_relu) {
    int wave  = (blockIdx.x * blockDim.x + threadIdx.x) >> 5;  // 0..24999
    int lane  = threadIdx.x & 31;
    int mtile = wave >> 3;          // 0..3124
    int ntile = wave & 7;           // 0..7
    int hl = lane & 15, hi = lane >> 4;

    int r = mtile * 16 + hl;        // A row this lane feeds
    int c = ntile * 16 + hl;        // output column this lane owns
    float s = 1.0f / fmaxf(deg[r], 1.0f);

    const float* arow = agg + (long)r * DIM;
    const float* xrow = xin + (long)r * DIM;

    v8f acc = {};
#pragma unroll
    for (int kk = 0; kk < DIM; kk += 32) {
        int ka = kk + hi * 16;      // A halves: 16 contiguous K for this half-wave
        int kb = kk + hi * 16;      // B halves: 16 contiguous K for this half-wave

        v16h a1 = load_frag_a(arow, ka, s);
        v16h b1 = *(const v16h*)(wl + (long)c * DIM + kb);   // B[k,c] = Wl[c,k]
        acc = __builtin_amdgcn_wmma_f32_16x16x32_f16(false, a1, false, b1,
                                                     (short)0, acc, false, false);

        v16h a2 = load_frag_a(xrow, ka, 1.0f);
        v16h b2 = *(const v16h*)(wr + (long)c * DIM + kb);
        acc = __builtin_amdgcn_wmma_f32_16x16x32_f16(false, a2, false, b2,
                                                     (short)0, acc, false, false);
    }

    float bv = bias[c];
#pragma unroll
    for (int g = 0; g < 8; ++g) {   // C/D layout: VGPR g -> row M = g + hi*8
        float v = acc[g] + bv;
        if (do_relu) v = fmaxf(v, 0.0f);
        out[(long)(mtile * 16 + hi * 8 + g) * DIM + c] = v;
    }
}

// ---------------------------------------------------------------- L2 normalize
__global__ void l2_normalize(float* __restrict__ out) {
    int tid  = blockIdx.x * blockDim.x + threadIdx.x;
    int node = tid >> 5;
    if (node >= N_NODES) return;
    int q = (tid & 31) * 4;
    float* p = out + (long)node * DIM + q;
    float4 v = *(const float4*)p;
    float ss = v.x * v.x + v.y * v.y + v.z * v.z + v.w * v.w;
#pragma unroll
    for (int off = 16; off >= 1; off >>= 1) ss += __shfl_xor(ss, off, 32);
    float sc = 1.0f / fmaxf(sqrtf(ss), 1e-12f);
    v.x *= sc; v.y *= sc; v.z *= sc; v.w *= sc;
    *(float4*)p = v;
}

// ---------------------------------------------------------------------- launch
extern "C" void kernel_launch(void* const* d_in, const int* in_sizes, int n_in,
                              void* d_out, int out_size, void* d_ws, size_t ws_size,
                              hipStream_t stream) {
    const float* x   = (const float*)d_in[0];
    const int*   ei  = (const int*)  d_in[1];
    const float* W1l = (const float*)d_in[2];
    const float* W1r = (const float*)d_in[3];
    const float* b1  = (const float*)d_in[4];
    const float* W2l = (const float*)d_in[5];
    const float* W2r = (const float*)d_in[6];
    const float* b2  = (const float*)d_in[7];
    const int* src = ei;
    const int* dst = ei + N_EDGES;

    char* ws = (char*)d_ws;
    float*    deg = (float*)   (ws + OFF_DEG);
    float*    agg = (float*)   (ws + OFF_AGG);
    float*    h1  = (float*)   (ws + OFF_H1);
    _Float16* wh  = (_Float16*)(ws + OFF_W);
    float*    out = (float*)d_out;

    // zero deg + agg (0xAA-poisoned workspace); capture-legal async memset
    hipMemsetAsync(ws, 0, OFF_H1, stream);
    cvt_weights<<<64, 256, 0, stream>>>(W1l, W1r, W2l, W2r, wh);
    deg_count<<<(N_EDGES + 255) / 256, 256, 0, stream>>>(dst, deg);

    // ---- layer 1
    scatter_add<<<(N_EDGES * 32) / 256, 256, 0, stream>>>(x, src, dst, agg);
    sage_gemm<<<3125, 256, 0, stream>>>(agg, deg, x, wh, wh + 16384, b1, h1, 1);

    // ---- layer 2 (reuse agg)
    hipMemsetAsync(ws + OFF_AGG, 0, (size_t)N_NODES * DIM * 4, stream);
    scatter_add<<<(N_EDGES * 32) / 256, 256, 0, stream>>>(h1, src, dst, agg);
    sage_gemm<<<3125, 256, 0, stream>>>(agg, deg, h1, wh + 32768, wh + 49152, b2, out, 0);

    // ---- row-wise L2 normalize in place
    l2_normalize<<<(N_NODES * 32) / 256, 256, 0, stream>>>(out);
}
